// NeuronLayer_53489522704546
// MI455X (gfx1250) — matrix operand order repaired and checked by reading, not compiled, not run on hardware
//
#include <hip/hip_runtime.h>
#include <math.h>

// ---------------------------------------------------------------------------
// Problem constants (fixed by the reference)
// ---------------------------------------------------------------------------
#define B_DIM      512
#define IN_DIM     2048
#define N_DIM      4096
#define BR_DIM     16
#define NTGT       4096

#define USE_TDM    1   // Tensor Data Mover staging (inline asm); 0 = sync copies

// vector types
typedef __attribute__((ext_vector_type(8)))  __bf16       v8bf;
typedef __attribute__((ext_vector_type(16))) __bf16       v16bf;
typedef __attribute__((ext_vector_type(8)))  float        v8f;
typedef __attribute__((ext_vector_type(4)))  unsigned int v4u;
typedef __attribute__((ext_vector_type(8)))  unsigned int v8u;

// ---------------------------------------------------------------------------
// fp32 -> bf16 (round to nearest even), raw u16 storage
// ---------------------------------------------------------------------------
__device__ __forceinline__ unsigned short f2bf(float f) {
    unsigned u = __float_as_uint(f);
    u += 0x7FFFu + ((u >> 16) & 1u);
    return (unsigned short)(u >> 16);
}

__global__ __launch_bounds__(256) void cvt_bf16_kernel(
    const float* __restrict__ in, unsigned short* __restrict__ out, int n) {
    int i = (blockIdx.x * 256 + threadIdx.x) * 4;
    if (i + 3 < n) {
        float4 v = *(const float4*)(in + i);
        out[i + 0] = f2bf(v.x);
        out[i + 1] = f2bf(v.y);
        out[i + 2] = f2bf(v.z);
        out[i + 3] = f2bf(v.w);
    }
}

// ---------------------------------------------------------------------------
// factors[n][br] = clip(att,0,1) * 0.9^delay
// ---------------------------------------------------------------------------
__global__ __launch_bounds__(256) void factors_kernel(
    const float* __restrict__ att, const int* __restrict__ delays,
    float* __restrict__ fac, int n) {
    int i = blockIdx.x * 256 + threadIdx.x;
    if (i < n) {
        float a = fminf(fmaxf(att[i], 0.0f), 1.0f);
        fac[i] = a * exp2f((float)delays[i] * -0.15200309344504997f);
    }
}

// ---------------------------------------------------------------------------
// GEMM: dend[m,n] = sum_k A[m,k]*W[n,k] + bias[n]   (bf16 in, f32 accum)
// Block: 256 thr / 8 waves (2M x 4N), block tile 64(M) x 128(N), K-tile 32.
// Staging: TDM tensor_load_to_lds, double-buffered, TENSORcnt-tracked.
// LDS pitch 40 bf16 (80B) produced by TDM pad (16 DWORD interval, 4 DWORD pad)
// -> conflict-free ds_load_b128 fragment fetches.
// ---------------------------------------------------------------------------
#define LDSPITCH    40
#define A_ROWS      64
#define B_ROWS      128
#define ABUF_BYTES  (A_ROWS * LDSPITCH * 2)   // 5120
#define BBUF_BYTES  (B_ROWS * LDSPITCH * 2)   // 10240

__device__ __forceinline__ unsigned lds_off(const void* p) {
    // generic -> LDS address space -> 32-bit byte offset
    return (unsigned)(unsigned long long)
        (__attribute__((address_space(3))) const void*)p;
}

#if USE_TDM
// Issue one TDM 2D tile load: rows x 32 bf16, row stride IN_DIM elements,
// LDS rows padded by 16B -> 80B pitch. Descriptor per CDNA5 ISA 8.3-8.5.
__device__ __forceinline__ void tdm_load_tile(const unsigned short* gbase,
                                              unsigned lds_addr, unsigned rows) {
    unsigned long long ga = (unsigned long long)gbase;
    // group0: count=1 | lds_addr | global_addr[56:0] | type=2
    v4u g0 = { 1u,
               lds_addr,
               (unsigned)ga,
               (unsigned)(ga >> 32) | (2u << 30) };
    // group1: data_size=2B, pad_enable, pad_interval=16DW, pad_amount=4DW
    //         tensor_dim0=2048, tensor_dim1=4096, tile_dim0=32, tile_dim1=rows
    //         tensor_dim0_stride=2048
    v8u g1 = { (1u << 16) | (1u << 20) | (3u << 22) | (3u << 25),
               (unsigned)(IN_DIM) << 16,          // tensor_dim0[15:0] @ [63:48]
               (unsigned)(N_DIM) << 16,           // tensor_dim1[15:0] @ [95:80]
               (32u << 16),                       // tile_dim0 @ [127:112]
               rows,                              // tile_dim1 @ [143:128]
               (unsigned)(IN_DIM),                // tensor_dim0_stride lo32
               0u, 0u };
    v4u gz = { 0u, 0u, 0u, 0u };                  // groups 2/3: unused dims
    asm volatile("tensor_load_to_lds %0, %1, %2, %3"
                 :: "s"(g0), "s"(g1), "s"(gz), "s"(gz)
                 : "memory");
}
#endif

__global__ __launch_bounds__(256) void gemm_wmma_kernel(
    const unsigned short* __restrict__ A,   // [B_DIM, IN_DIM] bf16
    const unsigned short* __restrict__ W,   // [N_DIM, IN_DIM] bf16
    const float* __restrict__ bias,         // [N_DIM]
    float* __restrict__ dend) {             // [B_DIM, N_DIM]
    __shared__ unsigned short As[2][A_ROWS * LDSPITCH];
    __shared__ unsigned short Bs[2][B_ROWS * LDSPITCH];

    const int tid  = threadIdx.x;
    const int lane = tid & 31;
    const int wave = tid >> 5;
    const int wm   = (wave & 1) * 32;   // 2 waves along M
    const int wn   = (wave >> 1) * 32;  // 4 waves along N
    const int m0   = blockIdx.y * 64;
    const int n0   = blockIdx.x * 128;
    const int lh   = lane >> 4;
    const int lr   = lane & 15;

    v8f acc[2][2];
    {
        v8f z = {};
#pragma unroll
        for (int i = 0; i < 2; ++i)
#pragma unroll
            for (int j = 0; j < 2; ++j) acc[i][j] = z;
    }

#if USE_TDM
    const unsigned asB = lds_off(&As[0][0]);
    const unsigned bsB = lds_off(&Bs[0][0]);
    if (wave == 0) {  // wave-uniform branch: EXEC stays all-1 per wave
        tdm_load_tile(A + (size_t)m0 * IN_DIM, asB, A_ROWS);
        tdm_load_tile(W + (size_t)n0 * IN_DIM, bsB, B_ROWS);
    }
#else
    const int ldRow = tid >> 1;
    const int ldSeg = (tid & 1) * 16;
#endif

    for (int kt = 0; kt < IN_DIM / 32; ++kt) {
        const int cur = kt & 1;
#if USE_TDM
        if (wave == 0) __builtin_amdgcn_s_wait_tensorcnt(0);
        __syncthreads();  // publish tile kt; all readers of buf 1-cur done
        if (wave == 0 && kt + 1 < IN_DIM / 32) {
            const int k0n = (kt + 1) * 32;
            tdm_load_tile(A + (size_t)m0 * IN_DIM + k0n,
                          asB + (1 - cur) * ABUF_BYTES, A_ROWS);
            tdm_load_tile(W + (size_t)n0 * IN_DIM + k0n,
                          bsB + (1 - cur) * BBUF_BYTES, B_ROWS);
        }
#else
        {
            const int k0 = kt * 32;
            if (ldRow < A_ROWS) {
                const v8bf* g = (const v8bf*)(A + (size_t)(m0 + ldRow) * IN_DIM + k0 + ldSeg);
                v8bf* s = (v8bf*)(&As[cur][ldRow * LDSPITCH + ldSeg]);
                s[0] = g[0];
                s[1] = g[1];
            }
            {
                const v8bf* g = (const v8bf*)(W + (size_t)(n0 + ldRow) * IN_DIM + k0 + ldSeg);
                v8bf* s = (v8bf*)(&Bs[cur][ldRow * LDSPITCH + ldSeg]);
                s[0] = g[0];
                s[1] = g[1];
            }
        }
        __syncthreads();
#endif
        const unsigned short* Asc = &As[cur][0];
        const unsigned short* Bsc = &Bs[cur][0];

        // A fragments: row lr; k = lh*8 + [0..7] and [16..23]
        v16bf af[2];
#pragma unroll
        for (int i = 0; i < 2; ++i) {
            const int r  = wm + i * 16 + lr;
            const int kb = lh * 8;
            union { v16bf v; v8bf h[2]; } u;
            u.h[0] = *(const v8bf*)(&Asc[r * LDSPITCH + kb]);
            u.h[1] = *(const v8bf*)(&Asc[r * LDSPITCH + kb + 16]);
            af[i] = u.v;
        }
        // B fragments: col lr; k = lh*8 + [0..7] and [16..23]
        v16bf bfr[2];
#pragma unroll
        for (int j = 0; j < 2; ++j) {
            const int r  = wn + j * 16 + lr;
            const int kb = lh * 8;
            union { v16bf v; v8bf h[2]; } u;
            u.h[0] = *(const v8bf*)(&Bsc[r * LDSPITCH + kb]);
            u.h[1] = *(const v8bf*)(&Bsc[r * LDSPITCH + kb + 16]);
            bfr[j] = u.v;
        }

#pragma unroll
        for (int i = 0; i < 2; ++i)
#pragma unroll
            for (int j = 0; j < 2; ++j)
                acc[i][j] = __builtin_amdgcn_wmma_f32_16x16x32_bf16(
                    false, af[i], false, bfr[j], (short)0, acc[i][j], false, false);

#if !USE_TDM
        __syncthreads();
#endif
    }

    // epilogue: C/D layout -> m = base + e + lh*8, n = base + lr; add bias
#pragma unroll
    for (int i = 0; i < 2; ++i) {
#pragma unroll
        for (int j = 0; j < 2; ++j) {
            const int n    = n0 + wn + j * 16 + lr;
            const float bv = bias[n];
#pragma unroll
            for (int e = 0; e < 8; ++e) {
                const int m = m0 + wm + i * 16 + lh * 8 + e;
                dend[(size_t)m * N_DIM + n] = acc[i][j][e] + bv;
            }
        }
    }
}

// ---------------------------------------------------------------------------
// LayerNorm + soma update (one block per batch row)
// ---------------------------------------------------------------------------
__global__ __launch_bounds__(256) void ln_soma_kernel(
    const float* __restrict__ dend, const float* __restrict__ mem,
    const float* __restrict__ refr, const float* __restrict__ gamma,
    const float* __restrict__ beta, const float* __restrict__ thr,
    const float* __restrict__ decay, float* __restrict__ spike_o,
    float* __restrict__ mem_o, float* __restrict__ ref_o,
    float* __restrict__ sig_o) {
    const int b = blockIdx.x;
    const int tid = threadIdx.x;
    const float* row = dend + (size_t)b * N_DIM;

    float s = 0.0f, q = 0.0f;
    for (int n = tid; n < N_DIM; n += 256) {
        float v = row[n];
        s += v;
        q += v * v;
    }
    __shared__ float sh[512];
    sh[tid] = s;
    sh[256 + tid] = q;
    __syncthreads();
    for (int off = 128; off > 0; off >>= 1) {
        if (tid < off) {
            sh[tid] += sh[tid + off];
            sh[256 + tid] += sh[256 + tid + off];
        }
        __syncthreads();
    }
    const float mu  = sh[0] * (1.0f / N_DIM);
    const float var = sh[256] * (1.0f / N_DIM) - mu * mu;
    const float inv = rsqrtf(var + 1e-5f);

    for (int n = tid; n < N_DIM; n += 256) {
        const size_t idx = (size_t)b * N_DIM + n;
        const float x  = (row[n] - mu) * inv * gamma[n] + beta[n];
        const float d  = fminf(fmaxf(decay[n], 0.0f), 1.0f);
        const float t  = fminf(fmaxf(thr[n], 0.5f), 2.0f);
        float nm       = d * mem[idx] + x;
        const float cf = (refr[idx] <= 0.0f) ? 1.0f : 0.0f;
        const float sp = ((nm >= t) ? 1.0f : 0.0f) * cf;
        nm            -= sp * t;
        float nr       = (sp > 0.5f) ? 2.0f : (refr[idx] - 1.0f);
        nr             = fmaxf(nr, 0.0f);
        const float sg = sp + 0.1f / (1.0f + __expf(-nm));
        spike_o[idx] = sp;
        mem_o[idx]   = nm;
        ref_o[idx]   = nr;
        sig_o[idx]   = sg;
    }
}

// ---------------------------------------------------------------------------
// Axon routing: per batch row, scatter-add into LDS (ds_add_f32), then one
// coalesced row write. No global atomics, no global zero-init.
// ---------------------------------------------------------------------------
__global__ __launch_bounds__(256) void axon_kernel(
    const float* __restrict__ sig, const float* __restrict__ fac,
    const int* __restrict__ tgt, float* __restrict__ out) {
    __shared__ float accum[NTGT];
    const int b = blockIdx.x;
    const int tid = threadIdx.x;
    for (int i = tid; i < NTGT; i += 256) accum[i] = 0.0f;
    __syncthreads();

    const float* srow = sig + (size_t)b * N_DIM;
    for (int n = tid; n < N_DIM; n += 256) {
        const float s = srow[n];
#pragma unroll
        for (int br = 0; br < BR_DIM; ++br) {
            const int t = tgt[n * BR_DIM + br];
            atomicAdd(&accum[t], s * fac[n * BR_DIM + br]);
        }
    }
    __syncthreads();
    float* orow = out + (size_t)b * NTGT;
    for (int i = tid; i < NTGT; i += 256) orow[i] = accum[i];
}

// ---------------------------------------------------------------------------
// launch
// ---------------------------------------------------------------------------
extern "C" void kernel_launch(void* const* d_in, const int* in_sizes, int n_in,
                              void* d_out, int out_size, void* d_ws, size_t ws_size,
                              hipStream_t stream) {
    const float* in_sig = (const float*)d_in[0];   // [B, IN_DIM]
    const float* mem    = (const float*)d_in[1];   // [B, N]
    const float* refr   = (const float*)d_in[2];   // [B, N]
    const float* W      = (const float*)d_in[3];   // [N, IN_DIM]
    const float* bias   = (const float*)d_in[4];   // [N]
    const float* gamma  = (const float*)d_in[5];   // [N]
    const float* beta   = (const float*)d_in[6];   // [N]
    const float* thr    = (const float*)d_in[7];   // [N]
    const float* decay  = (const float*)d_in[8];   // [N]
    const float* att    = (const float*)d_in[9];   // [N, BR]
    const int*   tgt    = (const int*)d_in[10];    // [N, BR]
    const int*   delays = (const int*)d_in[11];    // [N, BR]

    // workspace layout (bytes)
    char* ws = (char*)d_ws;
    unsigned short* wsA = (unsigned short*)(ws);                      //  2 MB bf16 A
    unsigned short* wsW = (unsigned short*)(ws + (2ull  << 20));      // 16 MB bf16 W
    float* dend = (float*)(ws + (18ull << 20));                       //  8 MB
    float* sig  = (float*)(ws + (26ull << 20));                       //  8 MB
    float* fac  = (float*)(ws + (34ull << 20));                       // 256 KB

    // output layout: (axon_out, spike, new_mem, new_ref) each [B, N]
    float* out    = (float*)d_out;
    float* axon   = out;
    float* spk_o  = out + (size_t)B_DIM * N_DIM;
    float* mem_o  = out + 2ull * B_DIM * N_DIM;
    float* ref_o  = out + 3ull * B_DIM * N_DIM;

    // 1) fp32 -> bf16 conversions
    {
        const int nA = B_DIM * IN_DIM;
        cvt_bf16_kernel<<<nA / 1024, 256, 0, stream>>>(in_sig, wsA, nA);
        const int nW = N_DIM * IN_DIM;
        cvt_bf16_kernel<<<nW / 1024, 256, 0, stream>>>(W, wsW, nW);
    }
    // 2) axon factors
    {
        const int nf = N_DIM * BR_DIM;
        factors_kernel<<<nf / 256, 256, 0, stream>>>(att, delays, fac, nf);
    }
    // 3) WMMA GEMM + bias (TDM staged, double buffered)
    {
        dim3 grid(N_DIM / 128, B_DIM / 64, 1);  // 32 x 8 = 256 blocks
        gemm_wmma_kernel<<<grid, 256, 0, stream>>>(wsA, wsW, bias, dend);
    }
    // 4) LayerNorm + soma
    ln_soma_kernel<<<B_DIM, 256, 0, stream>>>(dend, mem, refr, gamma, beta, thr,
                                              decay, spk_o, mem_o, ref_o, sig);
    // 5) axon scatter via LDS
    axon_kernel<<<B_DIM, 256, 0, stream>>>(sig, fac, tgt, axon);

    (void)in_sizes; (void)n_in; (void)out_size; (void)ws_size;
}
